// AxialRoPESelfAttention_15728170238566
// MI455X (gfx1250) — matrix operand order repaired and checked by reading, not compiled, not run on hardware
//
#include <hip/hip_runtime.h>

typedef __attribute__((ext_vector_type(16))) __bf16 v16bf;
typedef __attribute__((ext_vector_type(8)))  float  v8f;
typedef __attribute__((ext_vector_type(4)))  int    v4i;

#define B_    4
#define S_    2048
#define D_    1024
#define H_    16
#define HD_   64
#define M_TOT (B_*S_)   // 8192
#define NQKV  3072

#if defined(__HIP_DEVICE_COMPILE__) && __has_builtin(__builtin_amdgcn_global_load_async_to_lds_b128)
#define HAVE_ASYNC 1
#define AS1 __attribute__((address_space(1)))
#define AS3 __attribute__((address_space(3)))
#endif

union Frag16 { v16bf v; uint4 q[2]; };

__device__ __forceinline__ unsigned short f2bf(float f) {
    unsigned u = __float_as_uint(f);
    u += 0x7FFFu + ((u >> 16) & 1u);     // round-to-nearest-even
    return (unsigned short)(u >> 16);
}
__device__ __forceinline__ float bf2f(unsigned short s) {
    return __uint_as_float(((unsigned)s) << 16);
}

// 16-byte global -> LDS copy: async DMA (ASYNCcnt) on gfx1250, sync fallback.
__device__ __forceinline__ void cp16(const unsigned short* g, unsigned short* l) {
#ifdef HAVE_ASYNC
    __builtin_amdgcn_global_load_async_to_lds_b128((AS1 v4i*)g, (AS3 v4i*)l, 0, 0);
#else
    *(uint4*)l = *(const uint4*)g;
#endif
}

template<int N>
__device__ __forceinline__ void async_wait() {
#ifdef HAVE_ASYNC
#if __has_builtin(__builtin_amdgcn_s_wait_asynccnt)
    __builtin_amdgcn_s_wait_asynccnt(N);
#else
    asm volatile("s_wait_asynccnt %0" :: "i"(N) : "memory");
#endif
#endif
}

// ---------------------------------------------------------------------------
// Kernel 1: pack/convert fp32 inputs to bf16 working buffers.
// Weights are packed TRANSPOSED ([n][k]) so GEMM B-tiles are straight copies.
// ---------------------------------------------------------------------------
__global__ void convert_kernel(const float* __restrict__ x,
                               const float* __restrict__ Wq,
                               const float* __restrict__ Wk,
                               const float* __restrict__ Wv,
                               const float* __restrict__ bq,
                               const float* __restrict__ bk,
                               const float* __restrict__ bv,
                               const float* __restrict__ Wo,
                               unsigned short* __restrict__ Xbf,
                               unsigned short* __restrict__ WqkvT,
                               unsigned short* __restrict__ WoT,
                               float* __restrict__ bqkv) {
    long long i = (long long)blockIdx.x * blockDim.x + threadIdx.x;
    const long long nW  = (long long)D_ * NQKV;
    const long long nWo = (long long)D_ * D_;
    const long long nX  = (long long)M_TOT * D_;
    if (i < nW) {                         // WqkvT[n][k] = W*[k][n]
        int n = (int)(i / D_), k = (int)(i % D_);
        const float* W = (n < D_) ? Wq : (n < 2*D_) ? Wk : Wv;
        int nn = n & (D_ - 1);
        WqkvT[i] = f2bf(W[k * D_ + nn]);
    } else if (i < nW + nWo) {            // WoT[n][k] = Wo[k][n]
        long long j = i - nW;
        int n = (int)(j >> 10), k = (int)(j & 1023);
        WoT[j] = f2bf(Wo[k * D_ + n]);
    } else if (i < nW + nWo + nX) {
        long long j = i - nW - nWo;
        Xbf[j] = f2bf(x[j]);
    } else if (i < nW + nWo + nX + NQKV) {
        int j = (int)(i - nW - nWo - nX);
        bqkv[j] = (j < D_) ? bq[j] : (j < 2*D_) ? bk[j - D_] : bv[j - 2*D_];
    }
}

// ---------------------------------------------------------------------------
// Kernel 2/5: bf16 WMMA GEMM, 128x128 block tile, 8 waves of 32x64,
// double-buffered async LDS staging (A and B^T tiles are straight copies).
//   QKV=true : scatter epilogue into Q[B,H,S,64], K[B,H,S,64], V^T[B,H,64,S]
//   QKV=false: plain fp32 C = A@B + bias
// ---------------------------------------------------------------------------
template<bool QKV>
__global__ __launch_bounds__(256)
void gemm_kernel(const unsigned short* __restrict__ A,
                 const unsigned short* __restrict__ BmT,   // [n][k]
                 const float* __restrict__ bias,
                 unsigned short* __restrict__ Qbf,
                 unsigned short* __restrict__ Kbf,
                 unsigned short* __restrict__ Vt,
                 float* __restrict__ Cout) {
    __shared__ unsigned short As[2 * 128 * 40];   // [buf][m][k], stride 40
    __shared__ unsigned short Bs[2 * 128 * 40];   // [buf][n][k], stride 40

    const int tid  = threadIdx.x;
    const int lane = tid & 31;
    const int w    = tid >> 5;
    const int wm   = w >> 1, wn = w & 1;
    const int m0   = blockIdx.y * 128;
    const int n0   = blockIdx.x * 128;
    const int hl   = (lane < 16) ? 0 : 1;    // lane half
    const int l15  = lane & 15;

    // issue one 128x32 A tile + 128x32 B tile into LDS buffer `buf`
    // 512 chunks of 16B per matrix; 2 chunks per thread each => 4 async ops
    auto issueTile = [&](int t, int buf) {
        int k0 = t * 32;
        #pragma unroll
        for (int i = 0; i < 2; i++) {
            int idx = tid + i * 256;
            int r = idx >> 2, c = (idx & 3) * 8;
            cp16(&A  [(long long)(m0 + r) * D_ + k0 + c], &As[buf * 5120 + r * 40 + c]);
            cp16(&BmT[(long long)(n0 + r) * D_ + k0 + c], &Bs[buf * 5120 + r * 40 + c]);
        }
    };

    v8f zf;
    #pragma unroll
    for (int e = 0; e < 8; e++) zf[e] = 0.0f;
    v8f acc[2][4];
    #pragma unroll
    for (int a = 0; a < 2; a++)
        #pragma unroll
        for (int b = 0; b < 4; b++) acc[a][b] = zf;

    const int nT = D_ / 32;   // 32 k-tiles
    issueTile(0, 0);
    for (int t = 0; t < nT; t++) {
        int buf = t & 1;
        if (t + 1 < nT) { issueTile(t + 1, buf ^ 1); async_wait<4>(); }
        else            { async_wait<0>(); }
        __syncthreads();

        const unsigned short* as = &As[buf * 5120];
        const unsigned short* bs = &Bs[buf * 5120];
        // A fragment: lanes 0-15 row M, K=0-7 & 16-23; lanes 16-31 K=8-15 & 24-31
        Frag16 af[2], bf[4];
        #pragma unroll
        for (int fm = 0; fm < 2; fm++) {
            int row = wm * 32 + fm * 16 + l15;
            int kb = hl * 8;
            af[fm].q[0] = *(const uint4*)&as[row * 40 + kb];
            af[fm].q[1] = *(const uint4*)&as[row * 40 + kb + 16];
        }
        // B fragment: lanes 0-15 col N with K=0-7 & 16-23; lanes 16-31 K=8-15 & 24-31
        #pragma unroll
        for (int fn = 0; fn < 4; fn++) {
            int col = wn * 64 + fn * 16 + l15;
            int kb = hl * 8;
            bf[fn].q[0] = *(const uint4*)&bs[col * 40 + kb];
            bf[fn].q[1] = *(const uint4*)&bs[col * 40 + kb + 16];
        }
        #pragma unroll
        for (int fm = 0; fm < 2; fm++)
            #pragma unroll
            for (int fn = 0; fn < 4; fn++)
                acc[fm][fn] = __builtin_amdgcn_wmma_f32_16x16x32_bf16(
                    false, af[fm].v, false, bf[fn].v, (short)0, acc[fm][fn],
                    false, false);
        __syncthreads();
    }

    // Epilogue. C layout: VGPR r holds row (r + 8*half), col = lane%16.
    #pragma unroll
    for (int fm = 0; fm < 2; fm++) {
        #pragma unroll
        for (int fn = 0; fn < 4; fn++) {
            int coln = n0 + wn * 64 + fn * 16 + l15;
            float bv_ = bias[coln];
            #pragma unroll
            for (int r = 0; r < 8; r++) {
                int row = m0 + wm * 32 + fm * 16 + r + hl * 8;
                float v = acc[fm][fn][r] + bv_;
                if (QKV) {
                    int proj = coln >> 10;
                    int nn = coln & 1023;
                    int h = nn >> 6, d = nn & 63;
                    int b = row >> 11, s = row & (S_ - 1);
                    int bh = b * H_ + h;
                    if (proj == 0)
                        Qbf[((long long)(bh * S_ + s)) * HD_ + d] = f2bf(v);
                    else if (proj == 1)
                        Kbf[((long long)(bh * S_ + s)) * HD_ + d] = f2bf(v);
                    else
                        Vt[((long long)bh * HD_ + d) * S_ + s] = f2bf(v);
                } else {
                    Cout[(long long)row * D_ + coln] = v;
                }
            }
        }
    }
}

// ---------------------------------------------------------------------------
// Kernel 3: RoPE in-place on Q,K (one thread per rotary pair)
// ---------------------------------------------------------------------------
__global__ void rope_kernel(unsigned short* __restrict__ Qbf,
                            unsigned short* __restrict__ Kbf,
                            const int* __restrict__ rs) {
    int i = blockIdx.x * blockDim.x + threadIdx.x;   // B*H*S*32 threads
    int pair = i & 31;
    int s    = (i >> 5) & (S_ - 1);
    int bh   = i >> 16;
    int start = rs[0];
    if (s < start) return;
    float p   = (float)(s - start);
    float inv = __expf(-((float)pair * (1.0f / 32.0f)) * 9.210340371976184f);
    float f   = p * inv;
    float c = __cosf(f), sn = __sinf(f);
    long long base = ((long long)bh * S_ + s) * HD_ + pair * 2;
    {
        float e = bf2f(Qbf[base]), o = bf2f(Qbf[base + 1]);
        Qbf[base]     = f2bf(e * c - o * sn);
        Qbf[base + 1] = f2bf(o * c + e * sn);
    }
    {
        float e = bf2f(Kbf[base]), o = bf2f(Kbf[base + 1]);
        Kbf[base]     = f2bf(e * c - o * sn);
        Kbf[base + 1] = f2bf(o * c + e * sn);
    }
}

// ---------------------------------------------------------------------------
// Kernel 4: flash attention. 4 waves x 16 query rows; double-buffered 64-key
// K/V tiles DMA'd into LDS.
// ---------------------------------------------------------------------------
__global__ __launch_bounds__(128)
void attn_kernel(const unsigned short* __restrict__ Qbf,
                 const unsigned short* __restrict__ Kbf,
                 const unsigned short* __restrict__ Vt,
                 unsigned short* __restrict__ Abf) {
    __shared__ unsigned short Ks[2 * 64 * 72];  // [buf][key][d]
    __shared__ unsigned short Vs[2 * 64 * 72];  // [buf][d][key]  (from V^T)
    __shared__ unsigned short Ps[4 * 16 * 72];  // per-wave P staging

    const int tid  = threadIdx.x;
    const int lane = tid & 31;
    const int w    = tid >> 5;
    const int hl   = (lane < 16) ? 0 : 1;
    const int l15  = lane & 15;
    const int bh   = blockIdx.x >> 5;
    const int qt   = blockIdx.x & 31;
    const int qbase = qt * 64 + w * 16;
    const long long qkBase = (long long)bh * S_ * HD_;
    const long long vBase  = (long long)bh * HD_ * S_;

    // issue one 64x64 K tile + V tile: 512 16B chunks each, 4 per thread each
    auto issueKV = [&](int t, int buf) {
        int kbase = t * 64;
        #pragma unroll
        for (int i = 0; i < 4; i++) {
            int idx = tid + i * 128;
            int r = idx >> 3, c = (idx & 7) * 8;
            cp16(&Kbf[qkBase + (long long)(kbase + r) * HD_ + c], &Ks[buf * 4608 + r * 72 + c]);
            cp16(&Vt [vBase  + (long long)r * S_ + kbase + c],    &Vs[buf * 4608 + r * 72 + c]);
        }
    };

    // Q A-fragments (held for the whole pass): K-dim 64 -> 2 frags
    Frag16 qa[2];
    {
        const unsigned short* qp = Qbf + qkBase + (long long)(qbase + l15) * HD_;
        #pragma unroll
        for (int kk = 0; kk < 2; kk++) {
            int d = kk * 32 + hl * 8;
            qa[kk].q[0] = *(const uint4*)&qp[d];
            qa[kk].q[1] = *(const uint4*)&qp[d + 16];
        }
    }

    v8f zf;
    #pragma unroll
    for (int e = 0; e < 8; e++) zf[e] = 0.0f;
    v8f o[4];
    #pragma unroll
    for (int db = 0; db < 4; db++) o[db] = zf;
    float m[8], lsum[8];
    #pragma unroll
    for (int r = 0; r < 8; r++) { m[r] = -3.0e38f; lsum[r] = 0.0f; }

    issueKV(0, 0);
    for (int kt = 0; kt < 32; kt++) {
        int buf = kt & 1;
        if (kt + 1 < 32) { issueKV(kt + 1, buf ^ 1); async_wait<8>(); }
        else             { async_wait<0>(); }
        __syncthreads();

        const unsigned short* ks = &Ks[buf * 4608];
        const unsigned short* vs = &Vs[buf * 4608];

        // S = Q @ K^T  (16 x 64 scores)
        v8f s4[4];
        #pragma unroll
        for (int fn = 0; fn < 4; fn++) s4[fn] = zf;
        #pragma unroll
        for (int fn = 0; fn < 4; fn++) {
            int key = fn * 16 + l15;
            #pragma unroll
            for (int kk = 0; kk < 2; kk++) {
                Frag16 kb;
                int d = kk * 32 + hl * 8;
                kb.q[0] = *(const uint4*)&ks[key * 72 + d];
                kb.q[1] = *(const uint4*)&ks[key * 72 + d + 16];
                s4[fn] = __builtin_amdgcn_wmma_f32_16x16x32_bf16(
                    false, qa[kk].v, false, kb.v, (short)0, s4[fn], false, false);
            }
        }

        // online softmax; each C row spans 16 lanes of one half
        float corr[8];
        #pragma unroll
        for (int r = 0; r < 8; r++) {
            float mt = -3.0e38f;
            #pragma unroll
            for (int fn = 0; fn < 4; fn++) {
                s4[fn][r] *= 0.125f;                       // 1/sqrt(64)
                mt = fmaxf(mt, s4[fn][r]);
            }
            #pragma unroll
            for (int off = 8; off >= 1; off >>= 1)
                mt = fmaxf(mt, __shfl_xor(mt, off, 32));
            float mn = fmaxf(m[r], mt);
            corr[r] = __expf(m[r] - mn);
            m[r] = mn;
            float rs = 0.0f;
            #pragma unroll
            for (int fn = 0; fn < 4; fn++) {
                float pv = __expf(s4[fn][r] - mn);
                s4[fn][r] = pv;
                rs += pv;
            }
            #pragma unroll
            for (int off = 8; off >= 1; off >>= 1)
                rs += __shfl_xor(rs, off, 32);
            lsum[r] = lsum[r] * corr[r] + rs;
            #pragma unroll
            for (int db = 0; db < 4; db++) o[db][r] *= corr[r];
        }

        // P: C-layout -> A-layout via per-wave LDS staging (bf16)
        #pragma unroll
        for (int fn = 0; fn < 4; fn++)
            #pragma unroll
            for (int r = 0; r < 8; r++)
                Ps[w * 1152 + (r + hl * 8) * 72 + fn * 16 + l15] = f2bf(s4[fn][r]);
        __syncthreads();

        Frag16 pa[2];
        #pragma unroll
        for (int kk = 0; kk < 2; kk++) {
            int k = kk * 32 + hl * 8;
            pa[kk].q[0] = *(const uint4*)&Ps[w * 1152 + l15 * 72 + k];
            pa[kk].q[1] = *(const uint4*)&Ps[w * 1152 + l15 * 72 + k + 16];
        }
        // O += P @ V
        #pragma unroll
        for (int db = 0; db < 4; db++) {
            int d = db * 16 + l15;
            #pragma unroll
            for (int kk = 0; kk < 2; kk++) {
                Frag16 vb;
                int k = kk * 32 + hl * 8;
                vb.q[0] = *(const uint4*)&vs[d * 72 + k];
                vb.q[1] = *(const uint4*)&vs[d * 72 + k + 16];
                o[db] = __builtin_amdgcn_wmma_f32_16x16x32_bf16(
                    false, pa[kk].v, false, vb.v, (short)0, o[db], false, false);
            }
        }
        __syncthreads();
    }

    // normalize and write bf16 attention output [B,S,H*64]
    int b = bh >> 4, h = bh & 15;
    #pragma unroll
    for (int db = 0; db < 4; db++) {
        #pragma unroll
        for (int r = 0; r < 8; r++) {
            int row = qbase + r + hl * 8;
            int d = db * 16 + l15;
            float v = o[db][r] / lsum[r];
            Abf[((long long)(b * S_ + row)) * D_ + h * HD_ + d] = f2bf(v);
        }
    }
}

// ---------------------------------------------------------------------------
extern "C" void kernel_launch(void* const* d_in, const int* in_sizes, int n_in,
                              void* d_out, int out_size, void* d_ws, size_t ws_size,
                              hipStream_t stream) {
    const float* x  = (const float*)d_in[0];
    const float* Wq = (const float*)d_in[1];
    const float* bq = (const float*)d_in[2];
    const float* Wk = (const float*)d_in[3];
    const float* bk = (const float*)d_in[4];
    const float* Wv = (const float*)d_in[5];
    const float* bv = (const float*)d_in[6];
    const float* Wo = (const float*)d_in[7];
    const float* bo = (const float*)d_in[8];
    const int*   rs = (const int*)d_in[9];

    char* p = (char*)d_ws;
    auto alloc = [&](size_t bytes) {
        char* r = p;
        p += (bytes + 255) & ~(size_t)255;
        return r;
    };
    unsigned short* Xbf   = (unsigned short*)alloc((size_t)M_TOT * D_ * 2);
    unsigned short* WqkvT = (unsigned short*)alloc((size_t)D_ * NQKV * 2);
    unsigned short* WoT   = (unsigned short*)alloc((size_t)D_ * D_ * 2);
    float*          bqkv  = (float*)alloc((size_t)NQKV * 4);
    unsigned short* Qbf   = (unsigned short*)alloc((size_t)M_TOT * D_ * 2);
    unsigned short* Kbf   = (unsigned short*)alloc((size_t)M_TOT * D_ * 2);
    unsigned short* Vt    = (unsigned short*)alloc((size_t)M_TOT * D_ * 2);
    unsigned short* Abf   = (unsigned short*)alloc((size_t)M_TOT * D_ * 2);

    {
        long long tot = (long long)D_ * NQKV + (long long)D_ * D_ +
                        (long long)M_TOT * D_ + NQKV;
        int blocks = (int)((tot + 255) / 256);
        convert_kernel<<<blocks, 256, 0, stream>>>(x, Wq, Wk, Wv, bq, bk, bv, Wo,
                                                   Xbf, WqkvT, WoT, bqkv);
    }
    {
        dim3 grid(NQKV / 128, M_TOT / 128);
        gemm_kernel<true><<<grid, 256, 0, stream>>>(Xbf, WqkvT, bqkv,
                                                    Qbf, Kbf, Vt, nullptr);
    }
    rope_kernel<<<(B_ * H_ * S_ * 32) / 256, 256, 0, stream>>>(Qbf, Kbf, rs);
    attn_kernel<<<B_ * H_ * (S_ / 64), 128, 0, stream>>>(Qbf, Kbf, Vt, Abf);
    {
        dim3 grid(D_ / 128, M_TOT / 128);
        gemm_kernel<false><<<grid, 256, 0, stream>>>(Abf, WoT, bo,
                                                     nullptr, nullptr, nullptr,
                                                     (float*)d_out);
    }
}
